// WSDModel_52725018526322
// MI455X (gfx1250) — matrix operand order repaired and checked by reading, not compiled, not run on hardware
//
#include <hip/hip_runtime.h>
#include <hip/hip_bf16.h>

typedef __bf16 bf16;
typedef bf16  v16bf __attribute__((ext_vector_type(16)));
typedef bf16  v8bf  __attribute__((ext_vector_type(8)));
typedef float v8f   __attribute__((ext_vector_type(8)));

#define BATCH 8
#define NSEQ  512
#define DDIM  768
#define YDIM  30000

#if defined(__has_builtin)
#  if __has_builtin(__builtin_amdgcn_global_load_async_to_lds_b128)
#    define HAVE_ASYNC_LDS 1
#  endif
#endif
#ifndef HAVE_ASYNC_LDS
#  define HAVE_ASYNC_LDS 0
#endif

// GCC-style int4 vector to match the builtin's declared parameter type exactly
typedef int v4i_gcc __attribute__((vector_size(16)));
typedef __attribute__((address_space(1))) v4i_gcc Gv4i;   // global
typedef __attribute__((address_space(3))) v4i_gcc Lv4i;   // LDS

__device__ inline void async_copy16(void* lds_generic, const void* gsrc) {
#if HAVE_ASYNC_LDS
  // LDS generic address carries the byte offset in addr[31:0]
  unsigned lds_off = (unsigned)(size_t)lds_generic;
  __builtin_amdgcn_global_load_async_to_lds_b128(
      (Gv4i*)(size_t)gsrc, (Lv4i*)(size_t)lds_off, 0, 0);
#else
  *(v8bf*)lds_generic = *(const v8bf*)gsrc;               // sync fallback
#endif
}

__device__ inline void wait_async0() {
#if HAVE_ASYNC_LDS
#  if __has_builtin(__builtin_amdgcn_s_wait_asynccnt)
  __builtin_amdgcn_s_wait_asynccnt(0);
#  else
  asm volatile("s_wait_asynccnt 0x0" ::: "memory");
#  endif
#endif
}

__device__ inline v16bf cat8(v8bf lo, v8bf hi) {
  return __builtin_shufflevector(lo, hi, 0,1,2,3,4,5,6,7,8,9,10,11,12,13,14,15);
}

// ---------------------------------------------------------------------------
// Elementwise kernels (vectorized)
// ---------------------------------------------------------------------------
__global__ void gather_embed(const int* __restrict__ Ms, const float* __restrict__ Ev,
                             float* __restrict__ Xf, bf16* __restrict__ Xb) {
  long q = (long)blockIdx.x * blockDim.x + threadIdx.x;  // over (B*N*D)/4
  const long total4 = (long)BATCH * NSEQ * DDIM / 4;
  if (q >= total4) return;
  long i = q * 4;
  int d = (int)(i % DDIM);
  long t = i / DDIM;                                     // b*N + n
  int tok = Ms[t];
  float4 v = *(const float4*)&Ev[(long)tok * DDIM + d];
  *(float4*)&Xf[i] = v;
  bf16 b4[4] = {(bf16)v.x, (bf16)v.y, (bf16)v.z, (bf16)v.w};
  *(uint2*)&Xb[i] = *(uint2*)b4;
}

__global__ void f32_to_bf16_v8(const float* __restrict__ in, bf16* __restrict__ out, long n8) {
  long q = (long)blockIdx.x * blockDim.x + threadIdx.x;  // over n/8
  if (q >= n8) return;
  long i = q * 8;
  float4 a = *(const float4*)&in[i];
  float4 b = *(const float4*)&in[i + 4];
  v8bf o;
  o[0]=(bf16)a.x; o[1]=(bf16)a.y; o[2]=(bf16)a.z; o[3]=(bf16)a.w;
  o[4]=(bf16)b.x; o[5]=(bf16)b.y; o[6]=(bf16)b.z; o[7]=(bf16)b.w;
  *(v8bf*)&out[i] = o;
}

// masked + position-biased softmax over the key axis (length NSEQ = 512)
__global__ void softmax_attn(const float* __restrict__ S, const int* __restrict__ Ms,
                             float* __restrict__ Aout, bf16* __restrict__ Ab) {
  int row = blockIdx.x;                 // b*N + i
  int b = row >> 9;
  int i = row & (NSEQ - 1);
  int tid = threadIdx.x;
  const float* srow = S + (size_t)row * NSEQ;
  const int*   mrow = Ms + (size_t)b * NSEQ;

  int j0 = tid, j1 = tid + 256;
  float v0 = (mrow[j0] != 0) ? srow[j0] + (float)(j0 - i) : -1e12f;
  float v1 = (mrow[j1] != 0) ? srow[j1] + (float)(j1 - i) : -1e12f;

  __shared__ float red[256];
  red[tid] = fmaxf(v0, v1);
  __syncthreads();
  for (int s = 128; s > 0; s >>= 1) {
    if (tid < s) red[tid] = fmaxf(red[tid], red[tid + s]);
    __syncthreads();
  }
  float rowmax = red[0];
  __syncthreads();

  float e0 = __expf(v0 - rowmax), e1 = __expf(v1 - rowmax);
  red[tid] = e0 + e1;
  __syncthreads();
  for (int s = 128; s > 0; s >>= 1) {
    if (tid < s) red[tid] += red[tid + s];
    __syncthreads();
  }
  float inv = 1.0f / red[0];

  float a0 = e0 * inv, a1 = e1 * inv;
  size_t base = (size_t)row * NSEQ;
  Aout[base + j0] = a0;  Aout[base + j1] = a1;
  Ab[base + j0] = (bf16)a0;  Ab[base + j1] = (bf16)a1;
}

// H = LayerNorm(Qc + X) * gamma + beta, emitted as bf16 (D = 768, 3 elems/thread)
__global__ void add_ln(const float* __restrict__ Qc, const float* __restrict__ Xf,
                       const float* __restrict__ g, const float* __restrict__ bt,
                       bf16* __restrict__ Hb) {
  int row = blockIdx.x;
  int tid = threadIdx.x;
  const float* q = Qc + (size_t)row * DDIM;
  const float* x = Xf + (size_t)row * DDIM;
  float v[3];
  float s = 0.f;
#pragma unroll
  for (int t = 0; t < 3; ++t) { int d = tid + t * 256; v[t] = q[d] + x[d]; s += v[t]; }

  __shared__ float red[256];
  red[tid] = s; __syncthreads();
  for (int st = 128; st > 0; st >>= 1) {
    if (tid < st) red[tid] += red[tid + st];
    __syncthreads();
  }
  float mu = red[0] * (1.0f / DDIM);
  __syncthreads();

  float vs = 0.f;
#pragma unroll
  for (int t = 0; t < 3; ++t) { float dlt = v[t] - mu; vs += dlt * dlt; }
  red[tid] = vs; __syncthreads();
  for (int st = 128; st > 0; st >>= 1) {
    if (tid < st) red[tid] += red[tid + st];
    __syncthreads();
  }
  float inv = rsqrtf(red[0] * (1.0f / DDIM) + 1e-5f);
#pragma unroll
  for (int t = 0; t < 3; ++t) {
    int d = tid + t * 256;
    Hb[(size_t)row * DDIM + d] = (bf16)((v[t] - mu) * inv * g[d] + bt[d]);
  }
}

// ---------------------------------------------------------------------------
// bf16 WMMA GEMM: C[M,N] = A[M,K] * B, B either [K,N] or transposed-stored [N,K].
// 256 threads (8 waves), block tile 128x128x32, wave tile 32x64 (2x4 WMMA tiles).
// TRANS_B path: double-buffered LDS with GLOBAL_LOAD_ASYNC_TO_LDS_B128 pipeline.
// Requires: M % 128 == 0, K % 32 == 0 (true for all call sites); only N may be ragged.
// ---------------------------------------------------------------------------
template <bool TRANS_B, bool OUT_BF16>
__global__ __launch_bounds__(256)
void gemm_bf16_wmma(const bf16* __restrict__ A, const bf16* __restrict__ Bm,
                    float* __restrict__ Cf, bf16* __restrict__ Cb,
                    int M, int N, int K,
                    long long sA, long long sB, long long sC) {
  constexpr int BM = 128, BN = 128, BK = 32, LDT = 40;   // 80B row pitch, 16B-aligned
  constexpr bool PIPE = (HAVE_ASYNC_LDS != 0) && TRANS_B;
  __shared__ __align__(16) bf16 As[2][BM][LDT];
  __shared__ __align__(16) bf16 Bs[2][BN][LDT];          // stored N-major: Bs[.][n][k]

  const int tid    = threadIdx.x;
  const int wave   = tid >> 5;
  const int lane   = tid & 31;
  const int waveM  = wave & 3;      // 0..3 -> 32-row band
  const int waveN  = wave >> 2;     // 0..1 -> 64-col band
  const int lane15 = lane & 15;
  const int laneHi = lane >> 4;

  const int blockM = blockIdx.y * BM;
  const int blockN = blockIdx.x * BN;
  const bf16* Ab = A  + (long long)blockIdx.z * sA;
  const bf16* Bb = Bm + (long long)blockIdx.z * sB;

  v8f acc[2][4];
#pragma unroll
  for (int i = 0; i < 2; ++i)
#pragma unroll
    for (int j = 0; j < 4; ++j) acc[i][j] = (v8f)0.0f;

  // ---- tile staging ----
  auto stageA = [&](int kt, int buf) {          // 512 x 16B chunks, 2 per thread
    int k0 = kt * BK;
#pragma unroll
    for (int c = tid; c < (BM * BK) / 8; c += 256) {
      int m = c >> 2, kq = (c & 3) * 8;
      async_copy16(&As[buf][m][kq], &Ab[(long long)(blockM + m) * K + k0 + kq]);
    }
  };
  auto stageB_t = [&](int kt, int buf) {        // TRANS_B: rows of B are K-contiguous
    int k0 = kt * BK;
#pragma unroll
    for (int c = tid; c < (BN * BK) / 8; c += 256) {
      int n = c >> 2, kq = (c & 3) * 8;
      int gn = blockN + n;
      if (gn < N)                               // stale LDS only feeds masked-out cols
        async_copy16(&Bs[buf][n][kq], &Bb[(long long)gn * K + k0 + kq]);
    }
  };
  auto stageB_n = [&](int kt, int buf) {        // [K,N] layout: vector read + transpose
    int k0 = kt * BK;
    for (int c = tid; c < (BN / 8) * BK; c += 256) {
      int k = c >> 4, ng = (c & 15) * 8;
      v8bf v = *(const v8bf*)&Bb[(long long)(k0 + k) * N + blockN + ng];
#pragma unroll
      for (int j = 0; j < 8; ++j) Bs[buf][ng + j][k] = v[j];
    }
  };

  // ---- compute one BK slab from LDS buffer `buf` ----
  auto compute = [&](int buf) {
    v16bf afrag[2], bfrag[4];
#pragma unroll
    for (int tm = 0; tm < 2; ++tm) {
      int mrow = waveM * 32 + tm * 16 + lane15;
      v8bf a0 = *(const v8bf*)&As[buf][mrow][laneHi * 8];
      v8bf a1 = *(const v8bf*)&As[buf][mrow][16 + laneHi * 8];
      afrag[tm] = cat8(a0, a1);
    }
#pragma unroll
    for (int tn = 0; tn < 4; ++tn) {
      int ncol = waveN * 64 + tn * 16 + lane15;
      v8bf b0 = *(const v8bf*)&Bs[buf][ncol][laneHi * 8];
      v8bf b1 = *(const v8bf*)&Bs[buf][ncol][16 + laneHi * 8];
      bfrag[tn] = cat8(b0, b1);
    }
#pragma unroll
    for (int tm = 0; tm < 2; ++tm)
#pragma unroll
      for (int tn = 0; tn < 4; ++tn)
        acc[tm][tn] = __builtin_amdgcn_wmma_f32_16x16x32_bf16(
            false, afrag[tm], false, bfrag[tn], (short)0, acc[tm][tn], false, false);
  };

  const int nk = K / BK;
  if (PIPE) {
    // async double-buffered pipeline: load k+1 while computing k
    stageA(0, 0); stageB_t(0, 0);
    wait_async0();
    __syncthreads();
    int cur = 0;
    for (int kt = 0; kt < nk; ++kt) {
      if (kt + 1 < nk) { stageA(kt + 1, cur ^ 1); stageB_t(kt + 1, cur ^ 1); }
      compute(cur);
      wait_async0();
      __syncthreads();
      cur ^= 1;
    }
  } else {
    for (int kt = 0; kt < nk; ++kt) {
      stageA(kt, 0);                           // async (if available) or vector copy
      if (TRANS_B) stageB_t(kt, 0); else stageB_n(kt, 0);
      wait_async0();
      __syncthreads();
      compute(0);
      __syncthreads();
    }
  }

  // ---- epilogue: C VGPR r -> row = r + 8*laneHi, col = lane15 ----
#pragma unroll
  for (int tm = 0; tm < 2; ++tm) {
#pragma unroll
    for (int tn = 0; tn < 4; ++tn) {
      int row0 = blockM + waveM * 32 + tm * 16 + laneHi * 8;
      int col  = blockN + waveN * 64 + tn * 16 + lane15;
      if (col < N) {
#pragma unroll
        for (int r = 0; r < 8; ++r) {
          long long off = (long long)blockIdx.z * sC + (long long)(row0 + r) * N + col;
          if (OUT_BF16) Cb[off] = (bf16)acc[tm][tn][r];
          else          Cf[off] = acc[tm][tn][r];
        }
      }
    }
  }
}

// ---------------------------------------------------------------------------
// Host-side orchestration
// ---------------------------------------------------------------------------
extern "C" void kernel_launch(void* const* d_in, const int* in_sizes, int n_in,
                              void* d_out, int out_size, void* d_ws, size_t ws_size,
                              hipStream_t stream) {
  const int*   Ms   = (const int*)  d_in[0];
  const float* Ev   = (const float*)d_in[1];
  const float* Ey   = (const float*)d_in[2];
  const float* WA   = (const float*)d_in[3];
  const float* WO   = (const float*)d_in[4];
  const float* gam  = (const float*)d_in[5];
  const float* bet  = (const float*)d_in[6];

  float* y_out = (float*)d_out;                                   // [B,N,Y]
  float* A_out = y_out + (long long)BATCH * NSEQ * YDIM;          // [B,N,N]

  const long long BN_ = (long long)BATCH * NSEQ;                  // 4096 rows
  char* w = (char*)d_ws;
  auto carve = [&](size_t bytes) { char* p = w; w += (bytes + 255) & ~(size_t)255; return p; };

  float* Xf   = (float*)carve(BN_ * DDIM * 4);
  bf16*  Xb   = (bf16*) carve(BN_ * DDIM * 2);
  bf16*  WAb  = (bf16*) carve((size_t)DDIM * DDIM * 2);
  bf16*  WOb  = (bf16*) carve((size_t)DDIM * DDIM * 2);
  bf16*  Eyb  = (bf16*) carve((size_t)YDIM * DDIM * 2);
  bf16*  QWb  = (bf16*) carve(BN_ * DDIM * 2);
  float* Sf   = (float*)carve(BN_ * NSEQ * 4);
  bf16*  Abf  = (bf16*) carve(BN_ * NSEQ * 2);
  bf16*  XWob = (bf16*) carve(BN_ * DDIM * 2);
  float* Qc   = (float*)carve(BN_ * DDIM * 4);
  bf16*  Hb   = (bf16*) carve(BN_ * DDIM * 2);

  // 1) embedding gather + bf16 casts of weights / output embedding
  {
    long tot4 = BN_ * DDIM / 4;
    gather_embed<<<dim3((tot4 + 255) / 256), dim3(256), 0, stream>>>(Ms, Ev, Xf, Xb);
    long wn8 = (long)DDIM * DDIM / 8;
    f32_to_bf16_v8<<<dim3((wn8 + 255) / 256), dim3(256), 0, stream>>>(WA, WAb, wn8);
    f32_to_bf16_v8<<<dim3((wn8 + 255) / 256), dim3(256), 0, stream>>>(WO, WOb, wn8);
    long en8 = (long)YDIM * DDIM / 8;
    f32_to_bf16_v8<<<dim3((en8 + 255) / 256), dim3(256), 0, stream>>>(Ey, Eyb, en8);
  }

  // 2) QW = X @ W_A            [4096,768] = [4096,768] x [768,768]
  gemm_bf16_wmma<false, true><<<dim3(6, 32, 1), dim3(256), 0, stream>>>(
      Xb, WAb, nullptr, QWb, (int)BN_, DDIM, DDIM, 0, 0, 0);

  // 3) S = QW @ X^T  (batched) [512,512] = [512,768] x [768,512]
  gemm_bf16_wmma<true, false><<<dim3(4, 4, BATCH), dim3(256), 0, stream>>>(
      QWb, Xb, Sf, nullptr, NSEQ, NSEQ, DDIM,
      (long long)NSEQ * DDIM, (long long)NSEQ * DDIM, (long long)NSEQ * NSEQ);

  // 4) A = softmax(S + posbias, mask)
  softmax_attn<<<dim3((unsigned)BN_), dim3(256), 0, stream>>>(Sf, Ms, A_out, Abf);

  // 5) XWo = X @ W_O
  gemm_bf16_wmma<false, true><<<dim3(6, 32, 1), dim3(256), 0, stream>>>(
      Xb, WOb, nullptr, XWob, (int)BN_, DDIM, DDIM, 0, 0, 0);

  // 6) Qc = A @ XWo  (batched) [512,768] = [512,512] x [512,768]
  gemm_bf16_wmma<false, false><<<dim3(6, 4, BATCH), dim3(256), 0, stream>>>(
      Abf, XWob, Qc, nullptr, NSEQ, DDIM, NSEQ,
      (long long)NSEQ * NSEQ, (long long)NSEQ * DDIM, (long long)NSEQ * DDIM);

  // 7) H = LN(Qc + X)
  add_ln<<<dim3((unsigned)BN_), dim3(256), 0, stream>>>(Qc, Xf, gam, bet, Hb);

  // 8) y = H @ E_y^T           [4096,30000] = [4096,768] x [768,30000]
  gemm_bf16_wmma<true, false><<<dim3((YDIM + 127) / 128, 32, 1), dim3(256), 0, stream>>>(
      Hb, Eyb, y_out, nullptr, (int)BN_, YDIM, DDIM, 0, 0, 0);
}
